// SA_19413252178719
// MI455X (gfx1250) — matrix operand order repaired and checked by reading, not compiled, not run on hardware
//
#include <hip/hip_runtime.h>
#include <stdint.h>

// ---------------------------------------------------------------------------
// SAGAN self-attention for MI455X (gfx1250, wave32, WMMA bf16 -> f32 acc)
//   B=4, C=512, H=W=64 -> N=4096, C8=64
// Pipeline:
//   k_prep     : one-time fp32 -> bf16 conversion of [Wf;Wg;Wh] (655 KB)
//   k_proj     : f = Wf x + bf, g = Wg x + bg, hh = Wh x + bh   (WMMA, bf16 out)
//   k_stats    : per-row i: m_i = max_j f_i.g_j, s_i = sum_j exp(.-m_i) (WMMA)
//   k_attn_out : out[c,j] = gamma * sum_i hh[c,i] * exp(f_i.g_j - m_i)/s_i + x
//                j-tile = 32 so each hh fragment load feeds two WMMAs
// ---------------------------------------------------------------------------

typedef __attribute__((ext_vector_type(16))) __bf16 v16bf;
typedef __attribute__((ext_vector_type(8)))  float  v8f;

#define WMMA_BF16(a, b, c) \
  __builtin_amdgcn_wmma_f32_16x16x32_bf16(false, (a), false, (b), (short)0, (c), false, false)

namespace {
constexpr int BB  = 4;
constexpr int CC  = 512;
constexpr int NN  = 4096;   // 64*64
constexpr int CQ  = 64;     // C/8
constexpr int RW  = 640;    // concatenated weight rows: 64(f)+64(g)+512(hh)
}

union Frag {
  v16bf v;
  uint4 q[2];
  unsigned short h[16];
};

__device__ __forceinline__ unsigned short f2bf(float f) {
  unsigned int u = __float_as_uint(f);
  u += 0x7FFFu + ((u >> 16) & 1u);           // round-to-nearest-even
  return (unsigned short)(u >> 16);
}

__device__ __forceinline__ v8f vzero8() {
  v8f z = {0.f, 0.f, 0.f, 0.f, 0.f, 0.f, 0.f, 0.f};
  return z;
}

// ---------------------------------------------------------------------------
// Kernel 0: one-time weight conversion to bf16, concatenated [640][512].
// ---------------------------------------------------------------------------
__global__ __launch_bounds__(256) void k_prep(
    const float* __restrict__ Wf, const float* __restrict__ Wg,
    const float* __restrict__ Wh, unsigned short* __restrict__ wcat)
{
  const int idx = blockIdx.x * 256 + threadIdx.x;   // 0 .. 640*512-1
  const int row = idx >> 9;          // /512
  const int c   = idx & 511;
  float v;
  if (row < 64)        v = Wf[row * CC + c];
  else if (row < 128)  v = Wg[(row - 64) * CC + c];
  else                 v = Wh[(row - 128) * CC + c];
  wcat[idx] = f2bf(v);
}

// ---------------------------------------------------------------------------
// Kernel 1: projections. Grid (N/16, B), 128 threads (4 waves).
// ---------------------------------------------------------------------------
__global__ __launch_bounds__(128) void k_proj(
    const float* __restrict__ x,
    const unsigned short* __restrict__ wcat,   // [640][512] bf16
    const float* __restrict__ bf_, const float* __restrict__ bg,
    const float* __restrict__ bh,
    unsigned short* __restrict__ fq,   // [B][N][64] bf16
    unsigned short* __restrict__ gk,   // [B][N][64] bf16
    unsigned short* __restrict__ vv)   // [B][C][N] bf16
{
  __shared__ unsigned short xlds[16][40];   // [n][c], pad to 40 (80B rows, 16B aligned)

  const int b    = blockIdx.y;
  const int n0   = blockIdx.x * 16;
  const int tid  = threadIdx.x;
  const int wave = tid >> 5;
  const int lane = tid & 31;
  const int col  = lane & 15;
  const int kh   = lane >> 4;

  v8f acc[10];
#pragma unroll
  for (int t = 0; t < 10; ++t) acc[t] = vzero8();

  const float* xb = x + (size_t)b * CC * NN;
  const int sn = tid & 15;       // staging: n offset
  const int sc = tid >> 4;       // staging: c offset 0..7

  for (int ks = 0; ks < 16; ++ks) {
    const int k0 = ks * 32;
    __syncthreads();             // protect previous iteration's LDS reads
#pragma unroll
    for (int p = 0; p < 4; ++p) {
      const int c = k0 + p * 8 + sc;
      xlds[sn][p * 8 + sc] = f2bf(xb[(size_t)c * NN + n0 + sn]);
    }
    if (ks + 1 < 16) {           // prefetch next K-step's x tile (global_prefetch_b8)
      __builtin_prefetch(&xb[(size_t)(k0 + 32 + sc) * NN + n0 + sn], 0, 3);
    }
    __syncthreads();

    Frag bfr;                    // B fragment: col = pixel, K = channel (contig)
    bfr.q[0] = *(const uint4*)&xlds[col][kh * 16];
    bfr.q[1] = *(const uint4*)&xlds[col][kh * 16 + 8];

#pragma unroll
    for (int t = 0; t < 10; ++t) {
      const int ro = (wave * 10 + t) * 16;
      const unsigned short* wrow = wcat + (size_t)(ro + col) * CC;
      Frag afr;                  // A fragment: row = out-channel, contiguous K
      afr.q[0] = *(const uint4*)&wrow[k0 + kh * 16];
      afr.q[1] = *(const uint4*)&wrow[k0 + kh * 16 + 8];
      acc[t] = WMMA_BF16(afr.v, bfr.v, acc[t]);
    }
  }

  // Epilogue: add bias (fp32), write bf16.
  const int n = n0 + col;
#pragma unroll
  for (int t = 0; t < 10; ++t) {
    const int ro = (wave * 10 + t) * 16;
    if (ro < 128) {
      const float* bias    = (ro < 64) ? bf_ : bg;
      unsigned short* dst  = (ro < 64) ? fq : gk;
      const int ob = (ro & 63) + kh * 8;        // 8 contiguous out-channels
      Frag r;
#pragma unroll
      for (int rr = 0; rr < 8; ++rr) r.h[rr] = f2bf(acc[t][rr] + bias[ob + rr]);
      *(uint4*)&dst[((size_t)(b * NN + n)) * CQ + ob] = r.q[0];
    } else {
      const int cb = (ro - 128) + kh * 8;
#pragma unroll
      for (int rr = 0; rr < 8; ++rr) {
        const int c = cb + rr;
        vv[((size_t)(b * CC + c)) * NN + n] = f2bf(acc[t][rr] + bh[c]);
      }
    }
  }
}

// ---------------------------------------------------------------------------
// Kernel 2: per-row softmax stats via online max/sum-exp WMMA sweep.
// Grid (N/64, B), 128 threads; each wave owns 16 rows, sweeps all 4096 cols.
// ---------------------------------------------------------------------------
__global__ __launch_bounds__(128) void k_stats(
    const unsigned short* __restrict__ fq,
    const unsigned short* __restrict__ gk,
    float* __restrict__ mbuf, float* __restrict__ sbuf)
{
  const int b    = blockIdx.y;
  const int lane = threadIdx.x & 31;
  const int col  = lane & 15;
  const int kh   = lane >> 4;
  const int i0   = blockIdx.x * 64 + (threadIdx.x >> 5) * 16;

  Frag a0, a1;   // query rows, loop-invariant
  const unsigned short* fr = fq + ((size_t)(b * NN + i0 + col)) * CQ;
  a0.q[0] = *(const uint4*)&fr[kh * 16];
  a0.q[1] = *(const uint4*)&fr[kh * 16 + 8];
  a1.q[0] = *(const uint4*)&fr[32 + kh * 16];
  a1.q[1] = *(const uint4*)&fr[32 + kh * 16 + 8];

  float m[8], s[8];
#pragma unroll
  for (int r = 0; r < 8; ++r) { m[r] = -3.0e38f; s[r] = 0.f; }

  const unsigned short* gb = gk + (size_t)b * NN * CQ;
  for (int jt = 0; jt < NN / 16; ++jt) {
    const unsigned short* gr = gb + ((size_t)(jt * 16 + col)) * CQ;
    Frag b0, b1;
    b0.q[0] = *(const uint4*)&gr[kh * 16];
    b0.q[1] = *(const uint4*)&gr[kh * 16 + 8];
    b1.q[0] = *(const uint4*)&gr[32 + kh * 16];
    b1.q[1] = *(const uint4*)&gr[32 + kh * 16 + 8];
    v8f c = vzero8();
    c = WMMA_BF16(a0.v, b0.v, c);
    c = WMMA_BF16(a1.v, b1.v, c);
#pragma unroll
    for (int r = 0; r < 8; ++r) {
      const float val = c[r];
      const float mn  = fmaxf(m[r], val);
      s[r] = s[r] * __expf(m[r] - mn) + __expf(val - mn);
      m[r] = mn;
    }
  }

  // Reduce across the 16 lanes holding each row's column partials.
#pragma unroll
  for (int mask = 1; mask < 16; mask <<= 1) {
#pragma unroll
    for (int r = 0; r < 8; ++r) {
      const float mo = __shfl_xor(m[r], mask, 32);
      const float so = __shfl_xor(s[r], mask, 32);
      const float mn = fmaxf(m[r], mo);
      s[r] = s[r] * __expf(m[r] - mn) + so * __expf(mo - mn);
      m[r] = mn;
    }
  }
  if (col == 0) {
#pragma unroll
    for (int r = 0; r < 8; ++r) {
      const int i = i0 + r + 8 * kh;
      mbuf[b * NN + i] = m[r];
      sbuf[b * NN + i] = s[r];
    }
  }
}

// ---------------------------------------------------------------------------
// Kernel 3: fused attention-apply + output. Grid (N/32, B), 128 threads.
// j-tile = 32 columns. Per 64-row i-step each wave recomputes a 16x32 logit
// tile (2 WMMA pairs), normalizes to P (bf16) into LDS (transposed [j][i]);
// after a barrier all 4 waves run the value GEMM over their 128-channel
// slice: each hh A-fragment load is reused for both 16-column P sub-tiles.
// ---------------------------------------------------------------------------
__global__ __launch_bounds__(128) void k_attn_out(
    const unsigned short* __restrict__ fq,
    const unsigned short* __restrict__ gk,
    const unsigned short* __restrict__ vv,
    const float* __restrict__ mbuf, const float* __restrict__ sbuf,
    const float* __restrict__ x, const float* __restrict__ gamma,
    float* __restrict__ out)
{
  __shared__ unsigned short P[32][72];   // [j-local][i-local], pad 72 (144B rows)

  const int b    = blockIdx.y;
  const int j0   = blockIdx.x * 32;
  const int wave = threadIdx.x >> 5;
  const int lane = threadIdx.x & 31;
  const int col  = lane & 15;
  const int kh   = lane >> 4;

  // Key B-fragments for both column sub-tiles (loop-invariant).
  Frag kb[2][2];
#pragma unroll
  for (int jh = 0; jh < 2; ++jh) {
    const unsigned short* gr = gk + ((size_t)(b * NN + j0 + jh * 16 + col)) * CQ;
    kb[jh][0].q[0] = *(const uint4*)&gr[kh * 16];
    kb[jh][0].q[1] = *(const uint4*)&gr[kh * 16 + 8];
    kb[jh][1].q[0] = *(const uint4*)&gr[32 + kh * 16];
    kb[jh][1].q[1] = *(const uint4*)&gr[32 + kh * 16 + 8];
  }

  v8f acc[2][8];
#pragma unroll
  for (int jh = 0; jh < 2; ++jh)
#pragma unroll
    for (int t = 0; t < 8; ++t) acc[jh][t] = vzero8();

  const float* mb = mbuf + b * NN;
  const float* sb = sbuf + b * NN;

  for (int is = 0; is < NN / 64; ++is) {
    const int ib = is * 64 + wave * 16;
    // --- logit tiles (16 i x 2x16 j), K = 64 over query dim ---
    Frag a0, a1;
    const unsigned short* fr = fq + ((size_t)(b * NN + ib + col)) * CQ;
    a0.q[0] = *(const uint4*)&fr[kh * 16];
    a0.q[1] = *(const uint4*)&fr[kh * 16 + 8];
    a1.q[0] = *(const uint4*)&fr[32 + kh * 16];
    a1.q[1] = *(const uint4*)&fr[32 + kh * 16 + 8];

    float mrow[8], rsrow[8];     // softmax stats for this wave's 16 i-rows
#pragma unroll
    for (int r = 0; r < 8; ++r) {
      const int i = ib + r + 8 * kh;
      mrow[r]  = mb[i];
      rsrow[r] = 1.0f / sb[i];
    }

#pragma unroll
    for (int jh = 0; jh < 2; ++jh) {
      v8f lg = vzero8();
      lg = WMMA_BF16(a0.v, kb[jh][0].v, lg);
      lg = WMMA_BF16(a1.v, kb[jh][1].v, lg);
      Frag p;
#pragma unroll
      for (int r = 0; r < 8; ++r)
        p.h[r] = f2bf(__expf(lg[r] - mrow[r]) * rsrow[r]);
      *(uint4*)&P[jh * 16 + col][wave * 16 + kh * 8] = p.q[0];  // [j][i] store
    }
    __syncthreads();

    // --- value GEMM: acc[jh][c-tile] += hh[c, i-step] * P[jh] ---
    const int i0 = is * 64;
#pragma unroll
    for (int ct = 0; ct < 8; ++ct) {
      const int cb = wave * 128 + ct * 16;
      const unsigned short* vr = vv + ((size_t)(b * CC + cb + col)) * NN + i0;
#pragma unroll
      for (int ksv = 0; ksv < 2; ++ksv) {
        const int k0 = ksv * 32;
        Frag av;
        av.q[0] = *(const uint4*)&vr[k0 + kh * 16];
        av.q[1] = *(const uint4*)&vr[k0 + kh * 16 + 8];
#pragma unroll
        for (int jh = 0; jh < 2; ++jh) {       // reuse av across both P tiles
          Frag bv;
          bv.q[0] = *(const uint4*)&P[jh * 16 + col][k0 + kh * 16];
          bv.q[1] = *(const uint4*)&P[jh * 16 + col][k0 + kh * 16 + 8];
          acc[jh][ct] = WMMA_BF16(av.v, bv.v, acc[jh][ct]);
        }
      }
    }
    __syncthreads();   // P consumed; safe to overwrite next step
  }

  // --- epilogue: out = gamma * sa + x ---
  const float g = gamma[0];
#pragma unroll
  for (int jh = 0; jh < 2; ++jh) {
#pragma unroll
    for (int ct = 0; ct < 8; ++ct) {
      const int cb = wave * 128 + ct * 16 + kh * 8;
#pragma unroll
      for (int r = 0; r < 8; ++r) {
        const size_t idx = ((size_t)(b * CC + cb + r)) * NN + j0 + jh * 16 + col;
        out[idx] = g * acc[jh][ct][r] + x[idx];
      }
    }
  }
}

// ---------------------------------------------------------------------------
extern "C" void kernel_launch(void* const* d_in, const int* in_sizes, int n_in,
                              void* d_out, int out_size, void* d_ws, size_t ws_size,
                              hipStream_t stream) {
  (void)in_sizes; (void)n_in; (void)out_size; (void)ws_size;
  const float* x     = (const float*)d_in[0];
  const float* Wf    = (const float*)d_in[1];
  const float* bf_   = (const float*)d_in[2];
  const float* Wg    = (const float*)d_in[3];
  const float* bg    = (const float*)d_in[4];
  const float* Wh    = (const float*)d_in[5];
  const float* bh    = (const float*)d_in[6];
  const float* gamma = (const float*)d_in[7];
  float* out = (float*)d_out;

  // Workspace (~22 MB): fq, gk [B][N][64] bf16; vv [B][C][N] bf16;
  //                     wcat [640][512] bf16; m,s f32
  unsigned short* fq   = (unsigned short*)d_ws;
  unsigned short* gk   = fq + (size_t)BB * NN * CQ;
  unsigned short* vv   = gk + (size_t)BB * NN * CQ;
  unsigned short* wcat = vv + (size_t)BB * CC * NN;
  float*          mbuf = (float*)(wcat + (size_t)RW * CC);
  float*          sbuf = mbuf + (size_t)BB * NN;

  k_prep<<<(RW * CC) / 256, 256, 0, stream>>>(Wf, Wg, Wh, wcat);

  dim3 g1(NN / 16, BB);
  k_proj<<<g1, 128, 0, stream>>>(x, wcat, bf_, bg, bh, fq, gk, vv);

  dim3 g2(NN / 64, BB);
  k_stats<<<g2, 128, 0, stream>>>(fq, gk, mbuf, sbuf);

  dim3 g3(NN / 32, BB);
  k_attn_out<<<g3, 128, 0, stream>>>(fq, gk, vv, mbuf, sbuf, x, gamma, out);
}
